// EX_Module_3D_7670811590904
// MI455X (gfx1250) — matrix operand order repaired and checked by reading, not compiled, not run on hardware
//
#include <hip/hip_runtime.h>
#include <hip/hip_bf16.h>
#include <math.h>

// Problem constants (B,C,D,H,W = 4,64,32,64,64)
#define NB   4
#define NC   64
#define NCH  32
#define NSP  131072            // D*H*W

#define WAVES_PER_WG     8
#define THREADS          (WAVES_PER_WG * 32)
#define CHUNK_COLS       256                      // columns staged per TDM tile
#define ITERS_PER_WG     8
#define CHUNKS_PER_BATCH (NSP / CHUNK_COLS)       // 512
#define WGS_PER_BATCH    (CHUNKS_PER_BATCH / ITERS_PER_WG)  // 64
#define TILES_PER_CHUNK  (CHUNK_COLS / 16)        // 16
#define TILES_PER_WAVE   (TILES_PER_CHUNK / WAVES_PER_WG)   // 2

// workspace float layout
#define ACC_STRIDE 80
#define ACC_CTX    0
#define ACC_G      32
#define ACC_SE     64
#define ACC_SCA    65
#define AVG_OFF    (NB * ACC_STRIDE)       // 320
#define SATT_OFF   (AVG_OFF + NB * NCH)    // 448
#define CONST_OFF  (SATT_OFF + NB * NC)    // 704

typedef __attribute__((ext_vector_type(16))) __bf16 v16bf;
typedef __attribute__((ext_vector_type(8)))  float  v8f;
typedef __attribute__((ext_vector_type(4)))  unsigned int v4u;
typedef __attribute__((ext_vector_type(8)))  int    v8i;
typedef __attribute__((ext_vector_type(4)))  int    v4i;

// ---------------- wave32 reduction helpers ----------------
__device__ __forceinline__ float warp_sum16(float v) {
  v += __shfl_xor(v, 1, 32);
  v += __shfl_xor(v, 2, 32);
  v += __shfl_xor(v, 4, 32);
  v += __shfl_xor(v, 8, 32);
  return v;  // every lane in each 16-lane half holds the half-sum
}
__device__ __forceinline__ float warp_sum32(float v) {
  v = warp_sum16(v);
  v += __shfl_xor(v, 16, 32);
  return v;
}
__device__ __forceinline__ float warp_max32(float v) {
  #pragma unroll
  for (int m = 1; m < 32; m <<= 1) v = fmaxf(v, __shfl_xor(v, m, 32));
  return v;
}

// ---------------- WMMA helper ----------------
__device__ __forceinline__ v8f wmma_bf16(v16bf a, v16bf b, v8f c) {
  return __builtin_amdgcn_wmma_f32_16x16x32_bf16(false, a, false, b,
                                                 (short)0, c, false, false);
}

// A fragment (16x32 bf16) from row-major W[rows][64]; rows >= `rows` zero-pad.
__device__ __forceinline__ v16bf load_a_frag(const float* __restrict__ W,
                                             int mbase, int rows, int khalf,
                                             int lane) {
  const int row = mbase + (lane & 15);
  const int kA  = khalf * 32 + ((lane >> 4) & 1) * 8;
  v16bf a;
  if (row < rows) {
    const float* wr = W + (size_t)row * 64 + kA;
    #pragma unroll
    for (int i = 0; i < 8; ++i) a[i]     = (__bf16)wr[i];
    #pragma unroll
    for (int i = 0; i < 8; ++i) a[8 + i] = (__bf16)wr[16 + i];
  } else {
    #pragma unroll
    for (int i = 0; i < 16; ++i) a[i] = (__bf16)0.0f;
  }
  return a;
}

// B fragment (32x16 bf16) from the staged LDS tile x_lds[64][CHUNK_COLS].
__device__ __forceinline__ v16bf load_b_frag_lds(const float* bp, int col0,
                                                 int khalf, int lane) {
  const int col = col0 + (lane & 15);
  const int kb  = khalf * 32 + ((lane >> 4) & 1) * 8;
  const float* p = bp + kb * CHUNK_COLS + col;
  v16bf bfr;
  #pragma unroll
  for (int i = 0; i < 16; ++i) bfr[i] = (__bf16)p[(i + (i >= 8 ? 8 : 0)) * CHUNK_COLS];
  return bfr;
}

// ---------------- Tensor Data Mover: DMA a 64 x CHUNK_COLS fp32 tile ----------
// D# per CDNA5 ISA. 2-D tile; groups 2/3 (and the extra
// int32x8 group of the 6-arg builtin) are zero/unused.
__device__ __forceinline__ void tdm_load_tile(const float* gsrc,
                                              unsigned lds_byte_addr) {
  const unsigned long long ga = (unsigned long long)(uintptr_t)gsrc;
  v4u g0;
  g0[0] = 1u;                                   // count=1, user descriptor
  g0[1] = lds_byte_addr;                        // lds_addr [63:32]
  g0[2] = (unsigned)(ga & 0xFFFFFFFFu);         // global_addr [95:64]
  g0[3] = (unsigned)((ga >> 32) & 0x01FFFFFFu)  // global_addr [120:96]
          | (2u << 30);                         // type=2 ("image")
  v8i g1;
  g1[0] = (int)(2u << 16);                      // data_size=2 (4 bytes)
  g1[1] = (int)(((unsigned)NSP & 0xFFFFu) << 16);        // tensor_dim0 lo16
  g1[2] = (int)(((unsigned)NSP >> 16) | (64u << 16));    // dim0 hi16 | dim1 lo16
  g1[3] = (int)((unsigned)CHUNK_COLS << 16);    // tile_dim0 @ [127:112]
  g1[4] = (int)64;                              // tile_dim1=64 rows
  g1[5] = (int)NSP;                             // tensor_dim0_stride lo32
  g1[6] = 0;                                    // stride hi16 | dim1_stride lo16
  g1[7] = 0;
  v4i z4 = {0, 0, 0, 0};
  v8i z8 = {0, 0, 0, 0, 0, 0, 0, 0};
  __builtin_amdgcn_tensor_load_to_lds(g0, g1, z4, z4, z8, 0);
}

// ---------------- kernel 0: zero the accumulators ----------------
__global__ void init_ws_kernel(float* __restrict__ ws) {
  const int i = blockIdx.x * blockDim.x + threadIdx.x;
  if (i < NB * ACC_STRIDE) ws[i] = 0.0f;
}

// ---------------- pass 1: wqr/wvr/wql projections + reductions ----------------
__global__ void __launch_bounds__(THREADS) pass1_kernel(
    const float* __restrict__ x, const float* __restrict__ wqr,
    const float* __restrict__ wvr, const float* __restrict__ wql,
    float* __restrict__ ws) {
  __shared__ float lds_x[2][64 * CHUNK_COLS];   // 2 x 64 KB double buffer
  __shared__ float sctx[NCH];
  __shared__ float sgsum[NCH];
  __shared__ float ssum;
  if (threadIdx.x < NCH) { sctx[threadIdx.x] = 0.f; sgsum[threadIdx.x] = 0.f; }
  if (threadIdx.x == 0) ssum = 0.f;

  const int lane = threadIdx.x & 31;
  const int wid  = threadIdx.x >> 5;
  const int b    = blockIdx.y;
  const float* xb = x + (size_t)b * NC * NSP;

  v16bf a_vr[2][2], a_ql[2][2], a_qr[2];
  #pragma unroll
  for (int h = 0; h < 2; ++h) {
    a_vr[0][h] = load_a_frag(wvr,  0, 32, h, lane);
    a_vr[1][h] = load_a_frag(wvr, 16, 32, h, lane);
    a_ql[0][h] = load_a_frag(wql,  0, 32, h, lane);
    a_ql[1][h] = load_a_frag(wql, 16, 32, h, lane);
    a_qr[h]    = load_a_frag(wqr,  0,  1, h, lane);  // 1 real row, 15 zero
  }

  float accCtx[16], accG[16], accSum = 0.f;
  #pragma unroll
  for (int i = 0; i < 16; ++i) { accCtx[i] = 0.f; accG[i] = 0.f; }

  const int chunk0 = blockIdx.x * ITERS_PER_WG;
  if (wid == 0)
    tdm_load_tile(xb + (size_t)chunk0 * CHUNK_COLS,
                  (unsigned)(uintptr_t)&lds_x[0][0]);
  __syncthreads();  // also covers the sctx/sgsum init

  for (int it = 0; it < ITERS_PER_WG; ++it) {
    const int buf = it & 1;
    if (wid == 0) {
      if (it + 1 < ITERS_PER_WG) {
        tdm_load_tile(xb + (size_t)(chunk0 + it + 1) * CHUNK_COLS,
                      (unsigned)(uintptr_t)&lds_x[buf ^ 1][0]);
        __builtin_amdgcn_s_wait_tensorcnt((short)1);
      } else {
        __builtin_amdgcn_s_wait_tensorcnt((short)0);
      }
    }
    __syncthreads();                 // buf is ready for every wave
    const float* bp = &lds_x[buf][0];

    #pragma unroll
    for (int j = 0; j < TILES_PER_WAVE; ++j) {
      const int col0 = (wid * TILES_PER_WAVE + j) * 16;
      v16bf bf0 = load_b_frag_lds(bp, col0, 0, lane);
      v16bf bf1 = load_b_frag_lds(bp, col0, 1, lane);

      v8f cq  = {}; cq  = wmma_bf16(a_qr[0],    bf0, cq ); cq  = wmma_bf16(a_qr[1],    bf1, cq );
      v8f cv0 = {}; cv0 = wmma_bf16(a_vr[0][0], bf0, cv0); cv0 = wmma_bf16(a_vr[0][1], bf1, cv0);
      v8f cv1 = {}; cv1 = wmma_bf16(a_vr[1][0], bf0, cv1); cv1 = wmma_bf16(a_vr[1][1], bf1, cv1);
      v8f cl0 = {}; cl0 = wmma_bf16(a_ql[0][0], bf0, cl0); cl0 = wmma_bf16(a_ql[0][1], bf1, cl0);
      v8f cl1 = {}; cl1 = wmma_bf16(a_ql[1][0], bf0, cl1); cl1 = wmma_bf16(a_ql[1][1], bf1, cl1);

      // mask logits: row M=0 (wqr) -> cq[0], lanes 0-15. relu in [0,~3]: exp safe.
      float e  = __expf(fmaxf(cq[0], 0.f));
      float eo = __shfl_xor(e, 16, 32);
      float ecol = (lane < 16) ? e : eo;  // broadcast e[col] to both M-halves
      accSum += ecol;                     // column counted twice; halved later

      #pragma unroll
      for (int r = 0; r < 8; ++r) {
        accCtx[r]     += fmaxf(cv0[r], 0.f) * ecol;
        accCtx[8 + r] += fmaxf(cv1[r], 0.f) * ecol;
        accG[r]       += fmaxf(cl0[r], 0.f);
        accG[8 + r]   += fmaxf(cl1[r], 0.f);
      }
    }
    __syncthreads();                 // done reading buf before it is re-filled
  }

  // per-wave flush: butterfly over columns, then LDS atomics
  #pragma unroll
  for (int i = 0; i < 16; ++i) {
    float cs = warp_sum16(accCtx[i]);
    float gs = warp_sum16(accG[i]);
    const int m  = i >> 3, r = i & 7;
    const int ch = 16 * m + r + ((lane < 16) ? 0 : 8);
    if (lane == 0 || lane == 16) {
      atomicAdd(&sctx[ch], cs);
      atomicAdd(&sgsum[ch], gs);
    }
  }
  float se = warp_sum32(accSum) * 0.5f;
  if (lane == 0) atomicAdd(&ssum, se);
  __syncthreads();

  float* acc = ws + b * ACC_STRIDE;
  if (threadIdx.x < NCH) {
    atomicAdd(&acc[ACC_CTX + threadIdx.x], sctx[threadIdx.x]);
    atomicAdd(&acc[ACC_G   + threadIdx.x], sgsum[threadIdx.x]);
  }
  if (threadIdx.x == 0) atomicAdd(&acc[ACC_SE], ssum);
}

// ---------------- kernel: avg_x softmax + LN + spatial attention ----------------
__global__ void __launch_bounds__(32) finalize1_kernel(
    const float* __restrict__ wup, float* __restrict__ ws) {
  const int b = blockIdx.x;
  const int lane = threadIdx.x;
  float* acc = ws + b * ACC_STRIDE;

  float g  = acc[ACC_G + lane] * (1.0f / NSP);
  float gm = warp_max32(g);
  float eg = __expf(g - gm);
  float es = warp_sum32(eg);
  ws[AVG_OFF + b * NCH + lane] = eg / es;

  float ctx = acc[ACC_CTX + lane] / acc[ACC_SE];
  float mu  = warp_sum32(ctx) * (1.0f / NCH);
  float d   = ctx - mu;
  float var = warp_sum32(d * d) * (1.0f / NCH);
  float ln  = d * rsqrtf(var + 1e-5f);

  __shared__ float lns[NCH];
  lns[lane] = ln;
  __syncthreads();

  #pragma unroll
  for (int half = 0; half < 2; ++half) {
    const int c = lane + 32 * half;
    float t = 0.f;
    #pragma unroll
    for (int k = 0; k < NCH; ++k) t += wup[c * NCH + k] * lns[k];
    float sa = fminf(fmaxf((t + 3.0f) * (1.0f / 6.0f), 0.f), 1.f);  // HSigmoid
    ws[SATT_OFF + b * NC + c] = sa;
  }
}

// ---------------- pass 2: wvl projection -> channel_attn -> S_ca ----------------
__global__ void __launch_bounds__(THREADS) pass2_kernel(
    const float* __restrict__ x, const float* __restrict__ wvl,
    float* __restrict__ ws) {
  __shared__ float lds_x[2][64 * CHUNK_COLS];
  __shared__ float ssca;
  if (threadIdx.x == 0) ssca = 0.f;

  const int lane = threadIdx.x & 31;
  const int wid  = threadIdx.x >> 5;
  const int b    = blockIdx.y;
  const float* xb = x + (size_t)b * NC * NSP;

  v16bf a0[2], a1[2];
  #pragma unroll
  for (int h = 0; h < 2; ++h) {
    a0[h] = load_a_frag(wvl,  0, 32, h, lane);
    a1[h] = load_a_frag(wvl, 16, 32, h, lane);
  }
  float av[16];
  #pragma unroll
  for (int i = 0; i < 16; ++i) {
    const int m  = i >> 3, r = i & 7;
    const int ch = 16 * m + r + ((lane < 16) ? 0 : 8);
    av[i] = ws[AVG_OFF + b * NCH + ch];
  }

  float accS = 0.f;
  const int chunk0 = blockIdx.x * ITERS_PER_WG;
  if (wid == 0)
    tdm_load_tile(xb + (size_t)chunk0 * CHUNK_COLS,
                  (unsigned)(uintptr_t)&lds_x[0][0]);
  __syncthreads();

  for (int it = 0; it < ITERS_PER_WG; ++it) {
    const int buf = it & 1;
    if (wid == 0) {
      if (it + 1 < ITERS_PER_WG) {
        tdm_load_tile(xb + (size_t)(chunk0 + it + 1) * CHUNK_COLS,
                      (unsigned)(uintptr_t)&lds_x[buf ^ 1][0]);
        __builtin_amdgcn_s_wait_tensorcnt((short)1);
      } else {
        __builtin_amdgcn_s_wait_tensorcnt((short)0);
      }
    }
    __syncthreads();
    const float* bp = &lds_x[buf][0];

    #pragma unroll
    for (int j = 0; j < TILES_PER_WAVE; ++j) {
      const int col0 = (wid * TILES_PER_WAVE + j) * 16;
      v16bf bf0 = load_b_frag_lds(bp, col0, 0, lane);
      v16bf bf1 = load_b_frag_lds(bp, col0, 1, lane);
      v8f c0 = {}; c0 = wmma_bf16(a0[0], bf0, c0); c0 = wmma_bf16(a0[1], bf1, c0);
      v8f c1 = {}; c1 = wmma_bf16(a1[0], bf0, c1); c1 = wmma_bf16(a1[1], bf1, c1);
      float tp = 0.f;
      #pragma unroll
      for (int r = 0; r < 8; ++r) {
        tp += av[r]     * fmaxf(c0[r], 0.f);
        tp += av[8 + r] * fmaxf(c1[r], 0.f);
      }
      float tf = tp + __shfl_xor(tp, 16, 32);   // combine the two M-halves
      accS += 1.0f / (1.0f + __expf(-tf));      // sigmoid (dup across halves)
    }
    __syncthreads();
  }
  float tot = warp_sum32(accS) * 0.5f;
  if (lane == 0) atomicAdd(&ssca, tot);
  __syncthreads();
  if (threadIdx.x == 0) atomicAdd(&ws[b * ACC_STRIDE + ACC_SCA], ssca);
}

// ---------------- kernel: SK gate -> per-(b,c) constant ----------------
__global__ void __launch_bounds__(32) finalize2_kernel(
    const float* __restrict__ wsk1, const float* __restrict__ wsk2,
    float* __restrict__ ws) {
  const int b = blockIdx.x;
  const int lane = threadIdx.x;
  const float sca = ws[b * ACC_STRIDE + ACC_SCA];

  __shared__ float us[NC], a1s[NCH];
  #pragma unroll
  for (int half = 0; half < 2; ++half) {
    const int c = lane + 32 * half;
    const float sc = ws[SATT_OFF + b * NC + c];
    us[c] = (sc * sca + 1.0f) * (1.0f / NSP);   // u = mean(sequence+parallel)
  }
  __syncthreads();

  float t1 = 0.f;
  #pragma unroll
  for (int k = 0; k < NC; ++k) t1 += wsk1[lane * NC + k] * us[k];
  a1s[lane] = fmaxf(t1, 0.f);
  __syncthreads();

  float a2[2];
  #pragma unroll
  for (int half = 0; half < 2; ++half) {
    const int c = lane + 32 * half;
    float t = 0.f;
    #pragma unroll
    for (int k = 0; k < NCH; ++k) t += wsk2[c * NCH + k] * a1s[k];
    a2[half] = fmaxf(t, 0.f);
  }
  float m  = warp_max32(fmaxf(a2[0], a2[1]));
  float e0 = __expf(a2[0] - m), e1 = __expf(a2[1] - m);
  float s  = warp_sum32(e0 + e1);
  #pragma unroll
  for (int half = 0; half < 2; ++half) {
    const int c = lane + 32 * half;
    const float a  = (half ? e1 : e0) / s;
    const float sc = ws[SATT_OFF + b * NC + c];
    ws[CONST_OFF + b * NC + c] = a * sc * sca + 1.0f - a;  // sk.sum(-1)
  }
}

// ---------------- pass 3: out = const[b,c] + relu(wres @ x) ----------------
__global__ void __launch_bounds__(THREADS) pass3_kernel(
    const float* __restrict__ x, const float* __restrict__ wres,
    const float* __restrict__ ws, float* __restrict__ out) {
  __shared__ float lds_x[2][64 * CHUNK_COLS];
  const int lane = threadIdx.x & 31;
  const int wid  = threadIdx.x >> 5;
  const int b    = blockIdx.y;
  const float* xb = x   + (size_t)b * NC * NSP;
  float*       ob = out + (size_t)b * NC * NSP;

  v16bf aw[4][2];
  #pragma unroll
  for (int m = 0; m < 4; ++m) {
    aw[m][0] = load_a_frag(wres, 16 * m, 64, 0, lane);
    aw[m][1] = load_a_frag(wres, 16 * m, 64, 1, lane);
  }
  float cst[32];
  #pragma unroll
  for (int i = 0; i < 32; ++i) {
    const int m  = i >> 3, r = i & 7;
    const int ch = 16 * m + r + ((lane < 16) ? 0 : 8);
    cst[i] = ws[CONST_OFF + b * NC + ch];
  }

  const int chunk0 = blockIdx.x * ITERS_PER_WG;
  if (wid == 0)
    tdm_load_tile(xb + (size_t)chunk0 * CHUNK_COLS,
                  (unsigned)(uintptr_t)&lds_x[0][0]);
  __syncthreads();

  for (int it = 0; it < ITERS_PER_WG; ++it) {
    const int buf = it & 1;
    if (wid == 0) {
      if (it + 1 < ITERS_PER_WG) {
        tdm_load_tile(xb + (size_t)(chunk0 + it + 1) * CHUNK_COLS,
                      (unsigned)(uintptr_t)&lds_x[buf ^ 1][0]);
        __builtin_amdgcn_s_wait_tensorcnt((short)1);
      } else {
        __builtin_amdgcn_s_wait_tensorcnt((short)0);
      }
    }
    __syncthreads();
    const float* bp = &lds_x[buf][0];
    const int n0chunk = (chunk0 + it) * CHUNK_COLS;

    #pragma unroll
    for (int j = 0; j < TILES_PER_WAVE; ++j) {
      const int col0 = (wid * TILES_PER_WAVE + j) * 16;
      v16bf bf0 = load_b_frag_lds(bp, col0, 0, lane);
      v16bf bf1 = load_b_frag_lds(bp, col0, 1, lane);
      const int col = n0chunk + col0 + (lane & 15);
      #pragma unroll
      for (int m = 0; m < 4; ++m) {
        v8f c = {};
        c = wmma_bf16(aw[m][0], bf0, c);
        c = wmma_bf16(aw[m][1], bf1, c);
        const int rowbase = 16 * m + ((lane < 16) ? 0 : 8);
        #pragma unroll
        for (int r = 0; r < 8; ++r) {
          ob[(size_t)(rowbase + r) * NSP + col] = fmaxf(c[r], 0.f) + cst[m * 8 + r];
        }
      }
    }
    __syncthreads();
  }
}

// ---------------- launch ----------------
extern "C" void kernel_launch(void* const* d_in, const int* in_sizes, int n_in,
                              void* d_out, int out_size, void* d_ws, size_t ws_size,
                              hipStream_t stream) {
  const float* x    = (const float*)d_in[0];
  const float* wqr  = (const float*)d_in[1];
  const float* wvr  = (const float*)d_in[2];
  const float* wup  = (const float*)d_in[3];
  const float* wql  = (const float*)d_in[4];
  const float* wvl  = (const float*)d_in[5];
  const float* wsk1 = (const float*)d_in[6];
  const float* wsk2 = (const float*)d_in[7];
  const float* wres = (const float*)d_in[8];
  float* out = (float*)d_out;
  float* ws  = (float*)d_ws;

  (void)in_sizes; (void)n_in; (void)out_size; (void)ws_size;

  init_ws_kernel<<<1, 512, 0, stream>>>(ws);
  dim3 grid(WGS_PER_BATCH, NB);
  pass1_kernel<<<grid, THREADS, 0, stream>>>(x, wqr, wvr, wql, ws);
  finalize1_kernel<<<NB, 32, 0, stream>>>(wup, ws);
  pass2_kernel<<<grid, THREADS, 0, stream>>>(x, wvl, ws);
  finalize2_kernel<<<NB, 32, 0, stream>>>(wsk1, wsk2, ws);
  pass3_kernel<<<grid, THREADS, 0, stream>>>(x, wres, ws, out);
}
